// HGTPolicy_77919296684593
// MI455X (gfx1250) — compile-verified, matrix operand order checked by reading, not run on hardware
//
#include <hip/hip_runtime.h>
#include <math.h>

// ---------------- constants ----------------
#define N_OPS   20000
#define N_MACH  500
#define N_JOBS  2000
#define HID     128
#define HEADS   8
#define DHEAD   16
#define N_LAYERS 3
#define NPAIRS  50000
#define TOTAL_NODES 22500

typedef __attribute__((ext_vector_type(16))) _Float16 v16h;
typedef __attribute__((ext_vector_type(8)))  _Float16 v8h;
typedef __attribute__((ext_vector_type(8)))  float    v8f;

// ---------------- helpers ----------------
__device__ inline float atomicMaxF(float* addr, float value) {
    if (value >= 0.0f)
        return __int_as_float(atomicMax((int*)addr, __float_as_int(value)));
    else
        return __uint_as_float(atomicMin((unsigned int*)addr, __float_as_uint(value)));
}

// ---------------- generic fill ----------------
__global__ void fill_f32(float* p, float v, int n) {
    int i = blockIdx.x * 256 + threadIdx.x;
    if (i < n) p[i] = v;
}

// ---------------- weight packing: W[K][N] fp32 -> Wt[N][Kpad] f16 ----------------
__global__ void pack_wt(const float* __restrict__ W, _Float16* __restrict__ Wt,
                        int K, int N, int Kpad) {
    int i = blockIdx.x * 256 + threadIdx.x;
    if (i >= N * Kpad) return;
    int n = i / Kpad, kp = i % Kpad;
    float v = (kp < K) ? W[(size_t)kp * N + n] : 0.0f;
    Wt[i] = (_Float16)v;
}

// ---------------- embedding: Y[M][128] = X[M][F] @ w[F][128] + b ----------------
__global__ void embed_k(const float* __restrict__ X, const float* __restrict__ w,
                        const float* __restrict__ b, float* __restrict__ Y,
                        int M, int F) {
    int gid = blockIdx.x * 256 + threadIdx.x;
    if (gid >= M * HID) return;
    int m = gid >> 7, c = gid & 127;
    float acc = b[c];
    for (int f = 0; f < F; ++f) acc += X[m * F + f] * w[f * HID + c];
    Y[gid] = acc;
}

// ---------------- layernorm over last dim (128), in place ----------------
__global__ void ln128(float* __restrict__ Y, const float* __restrict__ g,
                      const float* __restrict__ b, int M) {
    __shared__ float red[HID];
    int m = blockIdx.x;
    int c = threadIdx.x;
    float v = Y[(size_t)m * HID + c];
    red[c] = v; __syncthreads();
    for (int s = 64; s > 0; s >>= 1) { if (c < s) red[c] += red[c + s]; __syncthreads(); }
    float mean = red[0] * (1.0f / HID); __syncthreads();
    float dv = v - mean;
    red[c] = dv * dv; __syncthreads();
    for (int s = 64; s > 0; s >>= 1) { if (c < s) red[c] += red[c + s]; __syncthreads(); }
    float var = red[0] * (1.0f / HID);
    Y[(size_t)m * HID + c] = dv * rsqrtf(var + 1e-5f) * g[c] + b[c];
}

// ---------------- temporal sin/cos embedding, padded to 32 cols ----------------
__global__ void temb_k(const float* __restrict__ t, float* __restrict__ temb, int E) {
    int gid = blockIdx.x * 256 + threadIdx.x;
    if (gid >= E * 16) return;
    int e = gid >> 4, j = gid & 15;
    float f = powf(1000.0f, -(float)(j >> 1) / 8.0f);  // 1/1000^(2i/16)
    float arg = t[e] * f;
    temb[e * 32 + j]      = (j & 1) ? cosf(arg) : sinf(arg);
    temb[e * 32 + 16 + j] = 0.0f;   // zero pad to K=32 for WMMA
}

// ---------------- scatter add pt into x_op[src] ----------------
__global__ void scat_add(const float* __restrict__ pt, const int* __restrict__ edge,
                         float* __restrict__ xop, int E) {
    int gid = blockIdx.x * 256 + threadIdx.x;
    if (gid >= E * HID) return;
    int e = gid >> 7, c = gid & 127;
    atomicAdd(&xop[(size_t)edge[e] * HID + c], pt[gid]);
}

// ---------------- WMMA GEMM: Y[M][N] = act(X[M][KPAD] @ Wt^T + bias) ----------------
// Wt packed as [N][KPAD] f16. One wave computes a 16x(16*NSUB) C block:
// A fragment loaded once per K-step and reused across NSUB column subtiles.
template<int KPAD, int NSUB>
__global__ void gemm_wmma(const float* __restrict__ X, const _Float16* __restrict__ Wt,
                          const float* __restrict__ bias, float* __restrict__ Y,
                          int M, int N, int act) {
    int lane = threadIdx.x & 31;
    int tm = blockIdx.x, tn = blockIdx.y;
    int r15 = lane & 15;
    int hi = lane >> 4;          // 0 or 1
    int khalf = hi * 8;
    int row = tm * 16 + r15;
    int rowc = row < M ? row : M - 1;
    int col0 = tn * (16 * NSUB) + r15;

    v8f acc[NSUB];
#pragma unroll
    for (int s = 0; s < NSUB; ++s) acc[s] = (v8f){};

#pragma unroll
    for (int k0 = 0; k0 < KPAD; k0 += 32) {
        // A fragment: lane holds row `rowc`, K = k0+khalf+{0..7} and +{16..23}
        const float4* ap = reinterpret_cast<const float4*>(X + (size_t)rowc * KPAD + k0 + khalf);
        float4 a0 = ap[0], a1 = ap[1], a2 = ap[4], a3 = ap[5];
        v16h A;
        A[0]=(_Float16)a0.x; A[1]=(_Float16)a0.y; A[2]=(_Float16)a0.z; A[3]=(_Float16)a0.w;
        A[4]=(_Float16)a1.x; A[5]=(_Float16)a1.y; A[6]=(_Float16)a1.z; A[7]=(_Float16)a1.w;
        A[8]=(_Float16)a2.x; A[9]=(_Float16)a2.y; A[10]=(_Float16)a2.z; A[11]=(_Float16)a2.w;
        A[12]=(_Float16)a3.x; A[13]=(_Float16)a3.y; A[14]=(_Float16)a3.z; A[15]=(_Float16)a3.w;
#pragma unroll
        for (int s = 0; s < NSUB; ++s) {
            // B fragment: lane holds column col0 + s*16, same K mapping
            const v8h* bp = reinterpret_cast<const v8h*>(
                Wt + (size_t)(col0 + s * 16) * KPAD + k0 + khalf);
            v8h b0 = bp[0], b1 = bp[2];
            v16h B;
#pragma unroll
            for (int i = 0; i < 8; ++i) { B[i] = b0[i]; B[8 + i] = b1[i]; }
            acc[s] = __builtin_amdgcn_wmma_f32_16x16x32_f16(false, A, false, B,
                                                            (short)0, acc[s], false, false);
        }
    }

    int rbase = tm * 16 + hi * 8;
    bool full = (tm * 16 + 16) <= M;   // uniform across the wave
#pragma unroll
    for (int s = 0; s < NSUB; ++s) {
        int col = col0 + s * 16;
        float bcol = bias ? bias[col] : 0.0f;
        float* yb = Y + (size_t)rbase * N + col;
        if (full) {
#pragma unroll
            for (int r = 0; r < 8; ++r) {
                float v = acc[s][r] + bcol;
                if (act == 1) v = v > 0.0f ? v : 0.0f;
                yb[(size_t)r * N] = v;
            }
        } else {
#pragma unroll
            for (int r = 0; r < 8; ++r) {
                if (rbase + r < M) {
                    float v = acc[s][r] + bcol;
                    if (act == 1) v = v > 0.0f ? v : 0.0f;
                    yb[(size_t)r * N] = v;
                }
            }
        }
    }
}

static inline void launch_gemm(const float* X, const _Float16* Wt, const float* bias,
                               float* Y, int M, int K, int N, int act, hipStream_t s) {
    // all N used here are multiples of 64 -> NSUB=4 (wave computes 16x64 of C)
    dim3 g((M + 15) / 16, N / 64), b(32);
    switch (K) {
        case 32:  gemm_wmma<32, 4><<<g, b, 0, s>>>(X, Wt, bias, Y, M, N, act); break;
        case 64:  gemm_wmma<64, 4><<<g, b, 0, s>>>(X, Wt, bias, Y, M, N, act); break;
        case 128: gemm_wmma<128, 4><<<g, b, 0, s>>>(X, Wt, bias, Y, M, N, act); break;
        case 256: gemm_wmma<256, 4><<<g, b, 0, s>>>(X, Wt, bias, Y, M, N, act); break;
    }
}

// ---------------- edge attention pass 1: logits + segment max ----------------
__global__ void edge_pass1(const float* __restrict__ Ksrc, const float* __restrict__ Qdst,
                           const int* __restrict__ edge, int E,
                           const float* __restrict__ a_rel, const float* __restrict__ p_rel,
                           float* __restrict__ abuf, float* __restrict__ smax) {
    __shared__ float sar[HEADS * 256];
    __shared__ float spr[HEADS];
    for (int i = threadIdx.x; i < HEADS * 256; i += 256) sar[i] = a_rel[i];
    if (threadIdx.x < HEADS) spr[threadIdx.x] = p_rel[threadIdx.x];
    __syncthreads();
    int gid = blockIdx.x * 256 + threadIdx.x;
    if (gid >= E * HEADS) return;
    int e = gid >> 3, h = gid & 7;
    int s = edge[e], d = edge[E + e];
    const float* kv = Ksrc + (size_t)s * HID + h * DHEAD;
    const float* qv = Qdst + (size_t)d * HID + h * DHEAD;
    const float* ar = sar + h * 256;
    float a = 0.0f;
#pragma unroll
    for (int j = 0; j < DHEAD; ++j) {
        float ks = 0.0f;
#pragma unroll
        for (int i = 0; i < DHEAD; ++i) ks += kv[i] * ar[i * 16 + j];
        a += qv[j] * ks;
    }
    a *= spr[h] * 0.25f;             // scale = 1/sqrt(16)
    abuf[gid] = a;
    atomicMaxF(&smax[d * HEADS + h], a);
}

// ---------------- edge attention pass 2: exp, sums, weighted V aggregate ----------------
__global__ void edge_pass2(const float* __restrict__ Vsrc, const int* __restrict__ edge, int E,
                           const float* __restrict__ m_rel, const float* __restrict__ abuf,
                           const float* __restrict__ smax, float* __restrict__ ssum,
                           float* __restrict__ agg) {
    __shared__ float smr[HEADS * 256];
    for (int i = threadIdx.x; i < HEADS * 256; i += 256) smr[i] = m_rel[i];
    __syncthreads();
    int gid = blockIdx.x * 256 + threadIdx.x;
    if (gid >= E * HEADS) return;
    int e = gid >> 3, h = gid & 7;
    int s = edge[e], d = edge[E + e];
    float m = smax[d * HEADS + h];
    float ex = expf(abuf[gid] - m);
    atomicAdd(&ssum[d * HEADS + h], ex);
    const float* vv = Vsrc + (size_t)s * HID + h * DHEAD;
    const float* mr = smr + h * 256;
#pragma unroll
    for (int j = 0; j < DHEAD; ++j) {
        float o = 0.0f;
#pragma unroll
        for (int i = 0; i < DHEAD; ++i) o += vv[i] * mr[i * 16 + j];
        atomicAdd(&agg[(size_t)d * HID + h * DHEAD + j], ex * o);
    }
}

// ---------------- normalize by softmax denom + exact gelu ----------------
__global__ void pass3_gelu(float* __restrict__ agg, const float* __restrict__ ssum, int Nn) {
    int gid = blockIdx.x * 256 + threadIdx.x;
    if (gid >= Nn * HID) return;
    int n = gid >> 7, c = gid & 127, h = c >> 4;
    float v = agg[gid] / (ssum[n * HEADS + h] + 1e-12f);
    agg[gid] = 0.5f * v * (1.0f + erff(v * 0.70710678f));
}

// ---------------- skip-gate combine ----------------
__global__ void combine_k(const float* __restrict__ o, const float* __restrict__ xold,
                          const float* __restrict__ skip, float* __restrict__ xn, int n) {
    int gid = blockIdx.x * 256 + threadIdx.x;
    if (gid >= n) return;
    float sk = 1.0f / (1.0f + expf(-skip[0]));
    xn[gid] = sk * o[gid] + (1.0f - sk) * xold[gid];
}

// ---------------- pair feature gather ----------------
__global__ void pair_gather(const float* __restrict__ xop, const float* __restrict__ xm,
                            const int* __restrict__ po, const int* __restrict__ pm,
                            float* __restrict__ pf) {
    int gid = blockIdx.x * 256 + threadIdx.x;
    if (gid >= NPAIRS * 256) return;
    int p = gid >> 8, c = gid & 255;
    pf[gid] = (c < HID) ? xop[(size_t)po[p] * HID + c]
                        : xm[(size_t)pm[p] * HID + (c - HID)];
}

// ---------------- final 64-dim dot for logits ----------------
__global__ void dot64(const float* __restrict__ H, const float* __restrict__ w,
                      const float* __restrict__ b, float* __restrict__ out, int M) {
    __shared__ float sw[64];
    if (threadIdx.x < 64) sw[threadIdx.x] = w[threadIdx.x];
    __syncthreads();
    int p = blockIdx.x * 256 + threadIdx.x;
    if (p >= M) return;
    float a = b[0];
    const float* h = H + (size_t)p * 64;
#pragma unroll
    for (int i = 0; i < 64; ++i) a += h[i] * sw[i];
    out[p] = a;
}

// ---------------- global mean accumulation ----------------
__global__ void mean_acc(const float* __restrict__ x, float* __restrict__ glob, int Nn) {
    int gid = blockIdx.x * 256 + threadIdx.x;
    if (gid >= Nn * HID) return;
    atomicAdd(&glob[gid & 127], x[gid] * (1.0f / (float)TOTAL_NODES));
}

// ---------------- value head (single block) ----------------
__global__ void value_head(const float* __restrict__ glob,
                           const float* __restrict__ w1, const float* __restrict__ b1,
                           const float* __restrict__ w2, const float* __restrict__ b2,
                           const float* __restrict__ w3, const float* __restrict__ b3,
                           float* __restrict__ out) {
    __shared__ float g[HID], h1[HID], h2[64];
    int c = threadIdx.x;
    g[c] = glob[c];
    __syncthreads();
    float a = b1[c];
    for (int i = 0; i < HID; ++i) a += g[i] * w1[i * HID + c];
    h1[c] = a > 0.0f ? a : 0.0f;
    __syncthreads();
    if (c < 64) {
        float a2 = b2[c];
        for (int i = 0; i < HID; ++i) a2 += h1[i] * w2[i * 64 + c];
        h2[c] = a2 > 0.0f ? a2 : 0.0f;
    }
    __syncthreads();
    if (c == 0) {
        float v = b3[0];
        for (int i = 0; i < 64; ++i) v += h2[i] * w3[i];
        out[0] = v;
    }
}

// ==================== host launcher ====================
extern "C" void kernel_launch(void* const* d_in, const int* in_sizes, int n_in,
                              void* d_out, int out_size, void* d_ws, size_t ws_size,
                              hipStream_t stream) {
    (void)in_sizes; (void)n_in; (void)out_size; (void)ws_size;
    auto F = [&](int i) { return (const float*)d_in[i]; };
    auto I = [&](int i) { return (const int*)d_in[i]; };

    // ---- input layout (top-level insertion order; nested dicts sorted-key) ----
    const float* op_x   = F(0);
    const float* mach_x = F(1);
    const float* job_x  = F(2);
    const float* t_attr = F(3);
    // edges sorted: assigned_to, belongs_to, can_process, contains,
    //               machine_precedes, on_machine, precedes, processes
    const int* ei[8]; for (int r = 0; r < 8; ++r) ei[r] = I(4 + r);
    const int relE[8]  = {20000, 20000, 60000, 20000, 20000, 20000, 20000, 20000};
    const int relST[8] = {0, 0, 1, 2, 0, 0, 0, 1};   // 0=op,1=machine,2=job
    const int relDT[8] = {1, 2, 0, 0, 0, 1, 0, 0};
    const int* pair_op = I(12);
    const int* pair_ma = I(13);
    // params: action_head, emb, layers[3], norm0, value_head
    const int PA = 14;            // ah0.b, ah0.w, ah1.b, ah1.w, ah2.b, ah2.w
    const int PE = PA + 6;        // emb: job.b,job.w, machine.b,machine.w, op.b,op.w
    const int PL = PE + 6;        // layers, 55 leaves each
    const int PN = PL + 3 * 55;   // norm0: job.b,job.g, machine.b,machine.g, op.b,op.g
    const int PV = PN + 6;        // vh0.b, vh0.w, vh1.b, vh1.w, vh2.b, vh2.w
    // per-layer group offsets (sorted keys within layer dict):
    const int G_ALIN = 0, G_AREL = 6, G_K = 14, G_MREL = 20, G_PREL = 28,
              G_Q = 36, G_SKIP = 42, G_TNORM = 45, G_TPROJ = 47, G_V = 49;
    const int toff[3] = {4, 2, 0};  // type (op,mach,job) -> offset in {job,machine,op} groups
    const int soff[3] = {2, 1, 0};  // skip scalar offsets

    const int NT[3] = {N_OPS, N_MACH, N_JOBS};
    float* out = (float*)d_out;

    // ---- workspace carving ----
    char* ws = (char*)d_ws;
    size_t off = 0;
    auto alloc = [&](size_t bytes) -> void* {
        void* p = ws + off;
        off = (off + bytes + 255) & ~(size_t)255;
        return p;
    };
    float *xcur[3], *xnew[3], *Kb[3], *Qb[3], *Vb[3], *aggb[3], *ob[3], *smax[3], *ssum[3];
    for (int t = 0; t < 3; ++t) {
        xcur[t] = (float*)alloc((size_t)NT[t] * HID * 4);
        xnew[t] = (float*)alloc((size_t)NT[t] * HID * 4);
        Kb[t]   = (float*)alloc((size_t)NT[t] * HID * 4);
        Qb[t]   = (float*)alloc((size_t)NT[t] * HID * 4);
        Vb[t]   = (float*)alloc((size_t)NT[t] * HID * 4);
        aggb[t] = (float*)alloc((size_t)NT[t] * HID * 4);
        ob[t]   = (float*)alloc((size_t)NT[t] * HID * 4);
        smax[t] = (float*)alloc((size_t)NT[t] * HEADS * 4);
        ssum[t] = (float*)alloc((size_t)NT[t] * HEADS * 4);
    }
    float* temb = (float*)alloc((size_t)20000 * 32 * 4);
    float* ptb  = (float*)alloc((size_t)20000 * HID * 4);
    float* abuf[8]; for (int r = 0; r < 8; ++r) abuf[r] = (float*)alloc((size_t)relE[r] * HEADS * 4);
    _Float16 *wtk[N_LAYERS][3], *wtq[N_LAYERS][3], *wtv[N_LAYERS][3], *wta[N_LAYERS][3], *wtt[N_LAYERS];
    for (int l = 0; l < N_LAYERS; ++l) {
        for (int t = 0; t < 3; ++t) {
            wtk[l][t] = (_Float16*)alloc(128 * 128 * 2);
            wtq[l][t] = (_Float16*)alloc(128 * 128 * 2);
            wtv[l][t] = (_Float16*)alloc(128 * 128 * 2);
            wta[l][t] = (_Float16*)alloc(128 * 128 * 2);
        }
        wtt[l] = (_Float16*)alloc(128 * 32 * 2);
    }
    _Float16* wt_ah0 = (_Float16*)alloc(128 * 256 * 2);
    _Float16* wt_ah1 = (_Float16*)alloc(64 * 128 * 2);
    float* pfeat = (float*)alloc((size_t)NPAIRS * 256 * 4);
    float* ph1   = (float*)alloc((size_t)NPAIRS * 128 * 4);
    float* ph2   = (float*)alloc((size_t)NPAIRS * 64 * 4);
    float* glob  = (float*)alloc(HID * 4);

    auto blocks = [](long n) { return (unsigned)((n + 255) / 256); };

    // ---- pack all weights (transpose + f16) ----
    for (int l = 0; l < N_LAYERS; ++l) {
        int L = PL + l * 55;
        for (int t = 0; t < 3; ++t) {
            pack_wt<<<blocks(128 * 128), 256, 0, stream>>>(F(L + G_K    + toff[t] + 1), wtk[l][t], 128, 128, 128);
            pack_wt<<<blocks(128 * 128), 256, 0, stream>>>(F(L + G_Q    + toff[t] + 1), wtq[l][t], 128, 128, 128);
            pack_wt<<<blocks(128 * 128), 256, 0, stream>>>(F(L + G_V    + toff[t] + 1), wtv[l][t], 128, 128, 128);
            pack_wt<<<blocks(128 * 128), 256, 0, stream>>>(F(L + G_ALIN + toff[t] + 1), wta[l][t], 128, 128, 128);
        }
        pack_wt<<<blocks(128 * 32), 256, 0, stream>>>(F(L + G_TPROJ + 1), wtt[l], 16, 128, 32);
    }
    pack_wt<<<blocks(128 * 256), 256, 0, stream>>>(F(PA + 1), wt_ah0, 256, 128, 256);
    pack_wt<<<blocks(64 * 128), 256, 0, stream>>>(F(PA + 3), wt_ah1, 128, 64, 128);

    // ---- embeddings + norm0 ----
    embed_k<<<blocks((long)N_OPS * HID), 256, 0, stream>>>(op_x,   F(PE + 5), F(PE + 4), xcur[0], N_OPS, 8);
    embed_k<<<blocks((long)N_MACH * HID), 256, 0, stream>>>(mach_x, F(PE + 3), F(PE + 2), xcur[1], N_MACH, 7);
    embed_k<<<blocks((long)N_JOBS * HID), 256, 0, stream>>>(job_x,  F(PE + 1), F(PE + 0), xcur[2], N_JOBS, 7);
    ln128<<<N_OPS, HID, 0, stream>>>(xcur[0], F(PN + 5), F(PN + 4), N_OPS);
    ln128<<<N_MACH, HID, 0, stream>>>(xcur[1], F(PN + 3), F(PN + 2), N_MACH);
    ln128<<<N_JOBS, HID, 0, stream>>>(xcur[2], F(PN + 1), F(PN + 0), N_JOBS);

    // ---- temporal sin/cos embedding (layer-invariant) ----
    temb_k<<<blocks((long)20000 * 16), 256, 0, stream>>>(t_attr, temb, 20000);

    // ---- HGT layers ----
    for (int l = 0; l < N_LAYERS; ++l) {
        int L = PL + l * 55;
        // relative temporal encoding: pt = LN(temb @ t_proj + b), scatter-add to op nodes
        launch_gemm(temb, wtt[l], F(L + G_TPROJ + 0), ptb, 20000, 32, HID, 0, stream);
        ln128<<<20000, HID, 0, stream>>>(ptb, F(L + G_TNORM + 1), F(L + G_TNORM + 0), 20000);
        scat_add<<<blocks((long)20000 * HID), 256, 0, stream>>>(ptb, ei[4], xcur[0], 20000);
        // K/Q/V projections (WMMA)
        for (int t = 0; t < 3; ++t) {
            launch_gemm(xcur[t], wtk[l][t], F(L + G_K + toff[t]), Kb[t], NT[t], 128, 128, 0, stream);
            launch_gemm(xcur[t], wtq[l][t], F(L + G_Q + toff[t]), Qb[t], NT[t], 128, 128, 0, stream);
            launch_gemm(xcur[t], wtv[l][t], F(L + G_V + toff[t]), Vb[t], NT[t], 128, 128, 0, stream);
        }
        // reset softmax scratch
        for (int t = 0; t < 3; ++t) {
            fill_f32<<<blocks((long)NT[t] * HEADS), 256, 0, stream>>>(smax[t], -3.0e38f, NT[t] * HEADS);
            fill_f32<<<blocks((long)NT[t] * HEADS), 256, 0, stream>>>(ssum[t], 0.0f, NT[t] * HEADS);
            fill_f32<<<blocks((long)NT[t] * HID), 256, 0, stream>>>(aggb[t], 0.0f, NT[t] * HID);
        }
        // edge attention: logits + segment max
        for (int r = 0; r < 8; ++r)
            edge_pass1<<<blocks((long)relE[r] * HEADS), 256, 0, stream>>>(
                Kb[relST[r]], Qb[relDT[r]], ei[r], relE[r],
                F(L + G_AREL + r), F(L + G_PREL + r), abuf[r], smax[relDT[r]]);
        // exp/sum + weighted V aggregation
        for (int r = 0; r < 8; ++r)
            edge_pass2<<<blocks((long)relE[r] * HEADS), 256, 0, stream>>>(
                Vb[relST[r]], ei[r], relE[r], F(L + G_MREL + r),
                abuf[r], smax[relDT[r]], ssum[relDT[r]], aggb[relDT[r]]);
        // normalize + gelu, a_lin projection, skip combine
        for (int t = 0; t < 3; ++t) {
            pass3_gelu<<<blocks((long)NT[t] * HID), 256, 0, stream>>>(aggb[t], ssum[t], NT[t]);
            launch_gemm(aggb[t], wta[l][t], F(L + G_ALIN + toff[t]), ob[t], NT[t], 128, 128, 0, stream);
            combine_k<<<blocks((long)NT[t] * HID), 256, 0, stream>>>(
                ob[t], xcur[t], F(L + G_SKIP + soff[t]), xnew[t], NT[t] * HID);
        }
        for (int t = 0; t < 3; ++t) { float* tmp = xcur[t]; xcur[t] = xnew[t]; xnew[t] = tmp; }
    }

    // ---- action head over 50k pairs ----
    pair_gather<<<blocks((long)NPAIRS * 256), 256, 0, stream>>>(xcur[0], xcur[1], pair_op, pair_ma, pfeat);
    launch_gemm(pfeat, wt_ah0, F(PA + 0), ph1, NPAIRS, 256, 128, 1, stream);
    launch_gemm(ph1, wt_ah1, F(PA + 2), ph2, NPAIRS, 128, 64, 1, stream);
    dot64<<<blocks(NPAIRS), 256, 0, stream>>>(ph2, F(PA + 5), F(PA + 4), out, NPAIRS);

    // ---- value head on global mean ----
    fill_f32<<<1, 256, 0, stream>>>(glob, 0.0f, HID);
    for (int t = 0; t < 3; ++t)
        mean_acc<<<blocks((long)NT[t] * HID), 256, 0, stream>>>(xcur[t], glob, NT[t]);
    value_head<<<1, HID, 0, stream>>>(glob, F(PV + 1), F(PV + 0), F(PV + 3), F(PV + 2),
                                      F(PV + 5), F(PV + 4), out + NPAIRS);
}
